// RE_GNN_71545565217169
// MI455X (gfx1250) — compile-verified, matrix operand order checked by reading, not run on hardware
//
#include <hip/hip_runtime.h>
#include <hip/hip_bf16.h>

// ---------------------------------------------------------------------------
// Shapes
#define Bb   16
#define Nn_  100
#define Ll   90
#define Dd   300
#define INn  256
#define Hh   256
// Pads (K dims multiples of 32, row pads multiples of 16)
#define LP   96      // L padded
#define DP   320     // D padded
#define NP   112     // N padded (rows)
#define SK   128     // N padded as K-dim for sim GEMMs
// ---------------------------------------------------------------------------

typedef __bf16 v16bf __attribute__((ext_vector_type(16)));
typedef float  v8f   __attribute__((ext_vector_type(8)));

union ABfrag { v16bf v; uint4 q[2]; };

static __device__ __forceinline__ __bf16 f2bf(float f) { return (__bf16)f; }
static __device__ __forceinline__ float  bf2f(__bf16 b) { return (float)b; }

static __device__ __forceinline__ v8f wmma_bf16(const ABfrag& a, const ABfrag& b, v8f c) {
    return __builtin_amdgcn_wmma_f32_16x16x32_bf16(false, a.v, false, b.v,
                                                   (short)0, c, false, false);
}

// --- CDNA5 async global->LDS copy (ASYNCcnt path) via inline asm ------------
// VDST operand is the 32-bit LDS byte address; flat pointers into the LDS
// aperture carry the LDS offset in their low 32 bits (ISA 10.2).
static __device__ __forceinline__ void copy16_g2l(const void* g, void* l) {
    unsigned loff = (unsigned)(unsigned long long)l;
    asm volatile("global_load_async_to_lds_b128 %0, %1, off"
                 :: "v"(loff), "v"(g) : "memory");
}
static __device__ __forceinline__ void async_wait0() {
    asm volatile("s_wait_asynccnt 0x0" ::: "memory");
}

// ---------------------------------------------------------------------------
// Small prep: Mt = Wk^T Wq  (mode 0), G = Wf Wv (mode 1) -> bf16 [320x320], 0-pad
__global__ void k_smallmm(const float* __restrict__ P, const float* __restrict__ Q,
                          __bf16* __restrict__ out, int mode) {
    int j = blockIdx.x * 16 + threadIdx.x;
    int i = blockIdx.y * 16 + threadIdx.y;
    float s = 0.f;
    if (i < Dd && j < Dd) {
        if (mode == 0) { for (int d = 0; d < Dd; ++d) s += P[d * Dd + i] * Q[d * Dd + j]; }
        else           { for (int d = 0; d < Dd; ++d) s += P[i * Dd + d] * Q[d * Dd + j]; }
    }
    out[i * DP + j] = f2bf(s);
}

// tq = Wq^T bk, tk = Wk^T bq, gb = 90*(Wf@bv + bf), c = bq.bk
__global__ void k_vec(const float* __restrict__ Wq, const float* __restrict__ bq,
                      const float* __restrict__ Wk, const float* __restrict__ bk,
                      const float* __restrict__ Wf, const float* __restrict__ bf_,
                      const float* __restrict__ bv,
                      float* tq, float* tk, float* gb, float* cbuf) {
    int t = threadIdx.x;               // 320 threads
    float s = 0.f, s2 = 0.f, s3 = 0.f;
    if (t < Dd) {
        for (int d = 0; d < Dd; ++d) { s += Wq[d * Dd + t] * bk[d]; s2 += Wk[d * Dd + t] * bq[d]; }
        float g = 0.f;
        for (int e = 0; e < Dd; ++e) g += Wf[t * Dd + e] * bv[e];
        s3 = 90.f * (g + bf_[t]);
    }
    tq[t] = s; tk[t] = s2; gb[t] = s3;
    if (t == 0) { float c = 0.f; for (int d = 0; d < Dd; ++d) c += bq[d] * bk[d]; cbuf[0] = c; }
}

// fp32 -> bf16 (weights already laid out as B^T row-major)
__global__ void k_wconvert(const float* __restrict__ in, __bf16* __restrict__ out, int n) {
    int i = blockIdx.x * 256 + threadIdx.x;
    if (i < n) out[i] = f2bf(in[i]);
}

// BatchNorm(eval) applied, bf16 output [1600,256]
__global__ void k_hbn(const float* __restrict__ h, const float* __restrict__ gamma,
                      const float* __restrict__ beta, __bf16* __restrict__ out) {
    int i = blockIdx.x * 256 + threadIdx.x;      // 1600*256
    int c = i & 255;
    out[i] = f2bf(h[i] * gamma[c] * rsqrtf(1.00001f) + beta[c]);
}

// ---------------------------------------------------------------------------
// Fused per-(b,n) attention-score kernel:
//   X(LDS,bf16) ; Y = X Mt^T (WMMA) ; S = Y X^T (WMMA) ; softmax rows ;
//   a = colsum(attn) ; xbar = a^T X ; itemsum = colsum(X)
// 1600 blocks x 192 threads (6 waves), dynamic LDS = 124032 B.
__global__ void k_score(const float* __restrict__ items, const __bf16* __restrict__ Mt,
                        const float* __restrict__ tq, const float* __restrict__ tk,
                        const float* __restrict__ cbuf,
                        float* __restrict__ itemsum, __bf16* __restrict__ xbar) {
    extern __shared__ char smem[];
    __bf16* Xs = (__bf16*)smem;                 // [96*320] bf16 = 61440 B
    __bf16* Ys = (__bf16*)(smem + 61440);       // [96*320] bf16 (dead after GEMM2)
    float*  Sf = (float*)(smem + 61440);        // [96*96] f32 overlays Ys
    float*  Us = (float*)(smem + 122880);       // [96]
    float*  Ws = (float*)(smem + 123264);       // [96]
    float*  As = (float*)(smem + 123648);       // [96]

    const int bn  = blockIdx.x;
    const int tid = threadIdx.x;
    const float* X = items + (long)bn * Ll * Dd;

    // stage X -> LDS bf16 (fp32->bf16 conversion in flight), zero-pad to 96x320
    for (int idx = tid; idx < LP * DP; idx += 192) {
        int l = idx / DP, e = idx - l * DP;
        float v = (l < Ll && e < Dd) ? X[l * Dd + e] : 0.f;
        Xs[idx] = f2bf(v);
    }
    __syncthreads();

    // itemsum (residual colsum) + rank-1 bias vectors u,w
    for (int e = tid; e < DP; e += 192) {
        float s = 0.f;
        for (int l = 0; l < Ll; ++l) s += bf2f(Xs[l * DP + e]);
        itemsum[(long)bn * DP + e] = s;
    }
    if (tid < LP) {
        float s = 0.f;
        for (int e = 0; e < DP; ++e) s += bf2f(Xs[tid * DP + e]) * tq[e];
        Us[tid] = s;
    } else {
        int l = tid - LP; float s = 0.f;
        for (int e = 0; e < DP; ++e) s += bf2f(Xs[l * DP + e]) * tk[e];
        Ws[l] = s;
    }
    __syncthreads();

    const int wave = tid >> 5, lane = tid & 31;
    const int arow = wave * 16 + (lane & 15);
    const int kb8  = (lane >> 4) << 3;

    // GEMM1: Y = X * Mt^T   (Mt rows are B^T rows)
    for (int ct = 0; ct < DP / 16; ++ct) {
        v8f acc = {0.f, 0.f, 0.f, 0.f, 0.f, 0.f, 0.f, 0.f};
        const __bf16* bp = Mt + (long)(ct * 16 + (lane & 15)) * DP + kb8;
        const __bf16* ap = Xs + arow * DP + kb8;
        for (int kk = 0; kk < DP / 32; ++kk) {
            ABfrag a, b;
            a.q[0] = *(const uint4*)(ap + kk * 32);
            a.q[1] = *(const uint4*)(ap + kk * 32 + 16);
            b.q[0] = *(const uint4*)(bp + kk * 32);
            b.q[1] = *(const uint4*)(bp + kk * 32 + 16);
            acc = wmma_bf16(a, b, acc);
        }
        int crow = wave * 16 + ((lane >> 4) << 3);
        int ccol = ct * 16 + (lane & 15);
        for (int v = 0; v < 8; ++v) Ys[(crow + v) * DP + ccol] = f2bf(acc[v]);
    }
    __syncthreads();

    // GEMM2: S = Y * X^T  (B^T rows = X rows)
    v8f zero = {0.f, 0.f, 0.f, 0.f, 0.f, 0.f, 0.f, 0.f};
    v8f sacc[6] = {zero, zero, zero, zero, zero, zero};
    for (int kk = 0; kk < DP / 32; ++kk) {
        ABfrag a;
        a.q[0] = *(const uint4*)(Ys + arow * DP + kb8 + kk * 32);
        a.q[1] = *(const uint4*)(Ys + arow * DP + kb8 + kk * 32 + 16);
        for (int mt = 0; mt < 6; ++mt) {
            ABfrag b;
            const __bf16* bp = Xs + (mt * 16 + (lane & 15)) * DP + kb8 + kk * 32;
            b.q[0] = *(const uint4*)(bp);
            b.q[1] = *(const uint4*)(bp + 16);
            sacc[mt] = wmma_bf16(a, b, sacc[mt]);
        }
    }
    __syncthreads();   // Ys dead; Sf may now overlay it

    const float cc = cbuf[0];
    int crow0 = wave * 16 + ((lane >> 4) << 3);
    for (int mt = 0; mt < 6; ++mt) {
        int col = mt * 16 + (lane & 15);
        for (int v = 0; v < 8; ++v) {
            int row = crow0 + v;
            Sf[row * LP + col] = 0.1f * (sacc[mt][v] + Us[row] + Ws[col] + cc);
        }
    }
    __syncthreads();

    // softmax over m<90 per row
    if (tid < Ll) {
        float mx = -3.4e38f;
        for (int m = 0; m < Ll; ++m) mx = fmaxf(mx, Sf[tid * LP + m]);
        float s = 0.f;
        for (int m = 0; m < Ll; ++m) { float e = __expf(Sf[tid * LP + m] - mx); Sf[tid * LP + m] = e; s += e; }
        float inv = 1.f / s;
        for (int m = 0; m < Ll; ++m) Sf[tid * LP + m] *= inv;
    }
    __syncthreads();
    if (tid < LP) {
        float s = 0.f;
        if (tid < Ll) for (int l = 0; l < Ll; ++l) s += Sf[l * LP + tid];
        As[tid] = s;
    }
    __syncthreads();
    // xbar = a^T X
    for (int e = tid; e < DP; e += 192) {
        float s = 0.f;
        for (int m = 0; m < Ll; ++m) s += As[m] * bf2f(Xs[m * DP + e]);
        xbar[(long)bn * DP + e] = f2bf(s);
    }
}

// ---------------------------------------------------------------------------
// row norms of hrev_bf (1792 rows x 320)
__global__ void k_norms(const __bf16* __restrict__ hrev, float* __restrict__ norms) {
    int r = blockIdx.x * 256 + threadIdx.x;
    if (r < Bb * NP) {
        float s = 0.f;
        for (int d = 0; d < DP; ++d) { float v = bf2f(hrev[(long)r * DP + d]); s += v * v; }
        norms[r] = sqrtf(s);
    }
}

// Per-batch: sim = hrev hrev^T / norms (WMMA), then two masked softmaxes.
// 16 blocks x 256 threads, dyn LDS = 122304 B.
__global__ void k_sim(const __bf16* __restrict__ hrev, const float* __restrict__ norms,
                      const int* __restrict__ adj,
                      __bf16* __restrict__ simin, __bf16* __restrict__ simout) {
    extern __shared__ char smem[];
    __bf16* Hs = (__bf16*)smem;                    // [112*320] = 71680 B
    float*  Sf = (float*)(smem + 71680);           // [112*112] = 50176 B
    float*  Ns = (float*)(smem + 71680 + 50176);   // [112]
    const int b = blockIdx.x, tid = threadIdx.x;
    const __bf16* H = hrev + (long)b * NP * DP;

    for (int idx = tid; idx < NP * DP / 8; idx += 256)
        copy16_g2l(H + idx * 8, Hs + idx * 8);     // async global->LDS (ASYNCcnt)
    if (tid < NP) Ns[tid] = norms[b * NP + tid];
    async_wait0();
    __syncthreads();

    const int wave = tid >> 5, lane = tid & 31;
    if (wave < 7) {
        const int arow = wave * 16 + (lane & 15);
        const int kb8  = (lane >> 4) << 3;
        v8f zero = {0.f, 0.f, 0.f, 0.f, 0.f, 0.f, 0.f, 0.f};
        v8f acc[7] = {zero, zero, zero, zero, zero, zero, zero};
        for (int kk = 0; kk < DP / 32; ++kk) {
            ABfrag a;
            a.q[0] = *(const uint4*)(Hs + arow * DP + kb8 + kk * 32);
            a.q[1] = *(const uint4*)(Hs + arow * DP + kb8 + kk * 32 + 16);
            for (int mt = 0; mt < 7; ++mt) {
                ABfrag bf;
                const __bf16* bp = Hs + (mt * 16 + (lane & 15)) * DP + kb8 + kk * 32;
                bf.q[0] = *(const uint4*)(bp);
                bf.q[1] = *(const uint4*)(bp + 16);
                acc[mt] = wmma_bf16(a, bf, acc[mt]);
            }
        }
        int crow0 = wave * 16 + ((lane >> 4) << 3);
        for (int mt = 0; mt < 7; ++mt) {
            int col = mt * 16 + (lane & 15);
            for (int v = 0; v < 8; ++v) {
                int row = crow0 + v;
                float den = fmaxf(Ns[row] * Ns[col], 1e-8f);
                Sf[row * NP + col] = acc[mt][v] / den;
            }
        }
    }
    __syncthreads();

    if (tid < Nn_) {
        const int n = tid;
        const int* ar = adj + ((long)b * Nn_ + n) * (2 * Nn_);
        const float NEG = -9e15f;
        // sim_in
        float mx = -3.4e38f;
        for (int m = 0; m < Nn_; ++m) { float v = (ar[m] > 0) ? Sf[n * NP + m] : NEG; mx = fmaxf(mx, v); }
        float s = 0.f;
        for (int m = 0; m < Nn_; ++m) { float v = (ar[m] > 0) ? Sf[n * NP + m] : NEG; s += __expf(v - mx); }
        float inv = 1.f / s;
        for (int m = 0; m < Nn_; ++m) {
            float v = (ar[m] > 0) ? Sf[n * NP + m] : NEG;
            simin[((long)b * NP + n) * SK + m] = f2bf(__expf(v - mx) * inv);
        }
        // sim_out
        mx = -3.4e38f;
        for (int m = 0; m < Nn_; ++m) { float v = (ar[Nn_ + m] > 0) ? Sf[n * NP + m] : NEG; mx = fmaxf(mx, v); }
        s = 0.f;
        for (int m = 0; m < Nn_; ++m) { float v = (ar[Nn_ + m] > 0) ? Sf[n * NP + m] : NEG; s += __expf(v - mx); }
        inv = 1.f / s;
        for (int m = 0; m < Nn_; ++m) {
            float v = (ar[Nn_ + m] > 0) ? Sf[n * NP + m] : NEG;
            simout[((long)b * NP + n) * SK + m] = f2bf(__expf(v - mx) * inv);
        }
    }
}

// ---------------------------------------------------------------------------
// Generic bf16 WMMA GEMM: C = A[MxK] * Bt[NnxK]^T (+bias)(+addend), batched via z.
// 128 threads (4 waves), block tile 64x64, wave tile 16x64.
// The 64xK B-tile is staged once into LDS (async) — all 4 waves share it.
// Requires Ncols % 64 == 0 (true for every launch below).
// rowmap: 0 identity; 1: r -> (r/100)*112 + r%100; 2: skip r%112>=100, compact.
// outmode: 0 f32; 1 bf16; 2 bf16 scatter-transpose ((r/100)*Nn+col)*128 + r%100.
__global__ void k_gemm(const __bf16* __restrict__ A, int lda, long sA,
                       const __bf16* __restrict__ Bt, int ldb, long sB,
                       float* __restrict__ Cf, __bf16* __restrict__ Cb, int ldc, long sC,
                       const float* __restrict__ bias, const float* __restrict__ addend, int ldadd,
                       int M, int Ncols, int K, int rowmap, int outmode) {
    extern __shared__ char gsm[];
    __bf16* Bs = (__bf16*)gsm;                 // [64][K] bf16
    const int wave = threadIdx.x >> 5, lane = threadIdx.x & 31;
    const long z = blockIdx.z;
    A  += z * sA;
    Bt += z * sB;
    const int row0 = blockIdx.y * 64 + wave * 16;
    const int col0 = blockIdx.x * 64;

    // stage B tile (64 rows x K) into LDS via async copies
    const int kch = K >> 3;                    // 16B chunks per row
    for (int idx = threadIdx.x; idx < 64 * kch; idx += 128) {
        int rowi = idx / kch, off = (idx - rowi * kch) * 8;
        copy16_g2l(Bt + (long)(col0 + rowi) * ldb + off, Bs + rowi * K + off);
    }

    int ar = row0 + (lane & 15); if (ar > M - 1) ar = M - 1;
    const int kb8 = (lane >> 4) << 3;
    const __bf16* aptr = A + (long)ar * lda + kb8;

    async_wait0();
    __syncthreads();

    v8f zero = {0.f, 0.f, 0.f, 0.f, 0.f, 0.f, 0.f, 0.f};
    v8f acc[4] = {zero, zero, zero, zero};

    const __bf16* bbase = Bs + (lane & 15) * K + kb8;
    const int nk = K >> 5;
    for (int kk = 0; kk < nk; ++kk) {
        ABfrag a;
        a.q[0] = *(const uint4*)(aptr);
        a.q[1] = *(const uint4*)(aptr + 16);
        __builtin_prefetch(aptr + 32, 0, 1);     // global_prefetch_b8
        aptr += 32;
        for (int t = 0; t < 4; ++t) {
            ABfrag b;
            const __bf16* bp = bbase + t * 16 * K + kk * 32;
            b.q[0] = *(const uint4*)(bp);
            b.q[1] = *(const uint4*)(bp + 16);
            acc[t] = wmma_bf16(a, b, acc[t]);
        }
    }

    const int rbase = row0 + ((lane >> 4) << 3);
    for (int t = 0; t < 4; ++t) {
        int col = col0 + t * 16 + (lane & 15);
        if (col >= Ncols) continue;
        float bv = bias ? bias[col] : 0.f;
        for (int v = 0; v < 8; ++v) {
            int r = rbase + v;
            if (r >= M) continue;
            float val = acc[t][v] + bv;
            if (addend) val += addend[(long)r * ldadd + col];
            if (outmode == 2) {
                long bb = r / Nn_, nn = r % Nn_;
                Cb[(bb * Ncols + col) * SK + nn] = f2bf(val);
                continue;
            }
            long orow;
            if (rowmap == 1)      orow = (long)(r / Nn_) * NP + (r % Nn_);
            else if (rowmap == 2) { int n = r % NP; if (n >= Nn_) continue; orow = (long)(r / NP) * Nn_ + n; }
            else                  orow = r;
            if (outmode == 0) Cf[z * sC + orow * ldc + col] = val;
            else              Cb[z * sC + orow * ldc + col] = f2bf(val);
        }
    }
}

// ---------------------------------------------------------------------------
extern "C" void kernel_launch(void* const* d_in, const int* in_sizes, int n_in,
                              void* d_out, int out_size, void* d_ws, size_t ws_size,
                              hipStream_t stream) {
    const float* h_local = (const float*)d_in[0];
    const float* items   = (const float*)d_in[1];
    const int*   adj     = (const int*)d_in[2];
    const float* gamma   = (const float*)d_in[4];
    const float* beta    = (const float*)d_in[5];
    const float* W_q = (const float*)d_in[6];  const float* b_q = (const float*)d_in[7];
    const float* W_k = (const float*)d_in[8];  const float* b_k = (const float*)d_in[9];
    const float* W_v = (const float*)d_in[10]; const float* b_v = (const float*)d_in[11];
    const float* W_f = (const float*)d_in[12]; const float* b_f = (const float*)d_in[13];
    const float* W_fc= (const float*)d_in[14]; const float* b_fc= (const float*)d_in[15];
    const float* W_in= (const float*)d_in[16]; const float* b_in= (const float*)d_in[17];
    const float* W_out=(const float*)d_in[18]; const float* b_out=(const float*)d_in[19];
    const float* b_iah=(const float*)d_in[20]; const float* b_oah=(const float*)d_in[21];
    const float* W_io= (const float*)d_in[22]; const float* b_io= (const float*)d_in[23];
    float* out = (float*)d_out;

    char* ws = (char*)d_ws;
    // workspace layout (bytes, 256-aligned); total ~11.8 MB
    __bf16* MT   = (__bf16*)(ws + 0);            // 320*320*2 = 204800
    __bf16* G    = (__bf16*)(ws + 204800);       // 204800
    float*  TQ   = (float*)(ws + 409600);        // 1280
    float*  TK   = (float*)(ws + 410880);        // 1280
    float*  GB   = (float*)(ws + 412160);        // 1280
    float*  CBUF = (float*)(ws + 413440);        // 256
    float*  ITEMSUM = (float*)(ws + 413696);     // 1600*320*4 = 2048000
    __bf16* XBAR = (__bf16*)(ws + 2461696);      // 1600*320*2 = 1024000
    __bf16* HREV = (__bf16*)(ws + 3485696);      // 16*112*320*2 = 1146880
    float*  NORMS= (float*)(ws + 4632576);       // 1792*4 = 7168
    __bf16* SIMIN= (__bf16*)(ws + 4639744);      // 16*112*128*2 = 458752
    __bf16* SIMOUT=(__bf16*)(ws + 5098496);      // 458752
    __bf16* HBN  = (__bf16*)(ws + 5557248);      // 1600*256*2 = 819200
    __bf16* WFC  = (__bf16*)(ws + 6376448);      // 131072
    __bf16* WIN  = (__bf16*)(ws + 6507520);      // 131072
    __bf16* WOUT = (__bf16*)(ws + 6638592);      // 131072
    __bf16* WIO  = (__bf16*)(ws + 6769664);      // 262144
    __bf16* HIDDEN=(__bf16*)(ws + 7031808);      // 819200
    __bf16* HINT = (__bf16*)(ws + 7851008);      // 16*256*128*2 = 1048576
    __bf16* HOUTT= (__bf16*)(ws + 8899584);      // 1048576
    __bf16* INIO = (__bf16*)(ws + 9948160);      // 16*112*512*2 = 1835008

    // zero padded buffers (pads must stay zero each call)
    (void)hipMemsetAsync(HREV,  0, 1146880, stream);
    (void)hipMemsetAsync(SIMIN, 0, 458752,  stream);
    (void)hipMemsetAsync(SIMOUT,0, 458752,  stream);
    (void)hipMemsetAsync(HINT,  0, 1048576, stream);
    (void)hipMemsetAsync(HOUTT, 0, 1048576, stream);
    (void)hipMemsetAsync(INIO,  0, 1835008, stream);

    // prep
    k_smallmm<<<dim3(20, 20), dim3(16, 16), 0, stream>>>(W_k, W_q, MT, 0); // Mt = Wk^T Wq
    k_smallmm<<<dim3(20, 20), dim3(16, 16), 0, stream>>>(W_f, W_v, G, 1);  // G = Wf Wv
    k_vec<<<1, 320, 0, stream>>>(W_q, b_q, W_k, b_k, W_f, b_f, b_v, TQ, TK, GB, CBUF);
    k_wconvert<<<(65536 + 255) / 256, 256, 0, stream>>>(W_fc, WFC, 65536);
    k_wconvert<<<(65536 + 255) / 256, 256, 0, stream>>>(W_in, WIN, 65536);
    k_wconvert<<<(65536 + 255) / 256, 256, 0, stream>>>(W_out, WOUT, 65536);
    k_wconvert<<<(131072 + 255) / 256, 256, 0, stream>>>(W_io, WIO, 131072);
    k_hbn<<<1600, 256, 0, stream>>>(h_local, gamma, beta, HBN);

    // fused attention-score path (dominant cost)
    k_score<<<1600, 192, 124032, stream>>>(items, MT, TQ, TK, CBUF, ITEMSUM, XBAR);

    // hidden = bn(h) @ Wfc^T + b_fc  -> bf16 [1600,256]
    k_gemm<<<dim3(4, 25, 1), 128, 128 * 256, stream>>>(HBN, 256, 0, WFC, 256, 0,
        nullptr, HIDDEN, 256, 0, b_fc, nullptr, 0, 1600, 256, 256, 0, 1);

    // hrev = itemsum + xbar @ G^T + gb  -> bf16 [b][112][320]
    k_gemm<<<dim3(5, 25, 1), 128, 128 * 320, stream>>>(XBAR, DP, 0, G, DP, 0,
        nullptr, HREV, DP, 0, GB, ITEMSUM, DP, 1600, DP, DP, 1, 1);

    k_norms<<<7, 256, 0, stream>>>(HREV, NORMS);
    k_sim<<<16, 256, 122304, stream>>>(HREV, NORMS, adj, SIMIN, SIMOUT);

    // hinT[b][256][128] = (hidden @ Win^T + b_in)^T  (scatter-transpose store)
    k_gemm<<<dim3(4, 25, 1), 128, 128 * 256, stream>>>(HIDDEN, 256, 0, WIN, 256, 0,
        nullptr, HINT, 0, 0, b_in, nullptr, 0, 1600, 256, 256, 0, 2);
    k_gemm<<<dim3(4, 25, 1), 128, 128 * 256, stream>>>(HIDDEN, 256, 0, WOUT, 256, 0,
        nullptr, HOUTT, 0, 0, b_out, nullptr, 0, 1600, 256, 256, 0, 2);

    // input_in / input_out: batched over z=16, K=128 (padded N)
    k_gemm<<<dim3(4, 2, 16), 128, 128 * 128, stream>>>(SIMIN, SK, (long)NP * SK,
        HINT, SK, (long)Hh * SK, nullptr, INIO, 512, (long)NP * 512,
        b_iah, nullptr, 0, NP, Hh, SK, 0, 1);
    k_gemm<<<dim3(4, 2, 16), 128, 128 * 128, stream>>>(SIMOUT, SK, (long)NP * SK,
        HOUTT, SK, (long)Hh * SK, nullptr, INIO + Hh, 512, (long)NP * 512,
        b_oah, nullptr, 0, NP, Hh, SK, 0, 1);

    // output = [in_in | in_out] @ Wio^T + b_io  (compact padded rows -> d_out)
    k_gemm<<<dim3(4, 28, 1), 128, 128 * 512, stream>>>(INIO, 512, 0, WIO, 512, 0,
        out, nullptr, Hh, 0, b_io, nullptr, 0, Bb * NP, Hh, 512, 2, 0);

    (void)in_sizes; (void)n_in; (void)out_size; (void)ws_size;
}
